// GraphNet_28862180229417
// MI455X (gfx1250) — matrix operand order repaired and checked by reading, not compiled, not run on hardware
//
#include <hip/hip_runtime.h>
#include <stdint.h>

typedef __attribute__((ext_vector_type(16))) __bf16 v16bf;
typedef __attribute__((ext_vector_type(8)))  float  v8f;

union Frag {
    uint32_t u[8];
    uint4    q[2];
    v16bf    v;
};

__device__ __forceinline__ uint16_t f2bf(float f) {
    union { float f; uint32_t u; } c; c.f = f;
    uint32_t u = c.u;
    u += 0x7FFFu + ((u >> 16) & 1u);   // round-to-nearest-even
    return (uint16_t)(u >> 16);
}
__device__ __forceinline__ float bf2f(uint32_t bits16) {
    union { uint32_t u; float f; } c; c.u = bits16 << 16;
    return c.f;
}

// ---------------------------------------------------------------- utilities
__global__ void k_zero(float* p, long n) {
    long i = (long)blockIdx.x * blockDim.x + threadIdx.x;
    if (i < n) p[i] = 0.f;
}

__global__ void k_cvt_bf16(const float* __restrict__ in, uint16_t* __restrict__ out, long n) {
    long i = (long)blockIdx.x * blockDim.x + threadIdx.x;
    if (i < n) out[i] = f2bf(in[i]);
}

// Pack weight matrix W[K][NCols] (row-major f32) into WMMA B-fragment order.
// Fragment tile (kt, nt): lane = n_local + 16*h; vgpr j holds rows k_local =
// 8h + 2j (+8 if j>=4) and +1, packed as two bf16 in one u32.
// ntMajor=0: linear tile index t = kt*NT + nt  (register-streamed weights)
// ntMajor=1: linear tile index t = nt*KT + kt  (contiguous 8KB per-nt slab for LDS staging)
__global__ void k_pack_w(const float* __restrict__ W, uint32_t* __restrict__ Wp,
                         int K, int NCols, int ntMajor) {
    int  KT = K >> 5;
    int  NT = NCols >> 4;
    long total = (long)KT * NT * 256;
    long i = (long)blockIdx.x * blockDim.x + threadIdx.x;
    if (i >= total) return;
    int  j    = (int)(i & 7);
    int  lane = (int)((i >> 3) & 31);
    long t    = i >> 8;
    int  nt, kt;
    if (ntMajor) { kt = (int)(t % KT); nt = (int)(t / KT); }
    else         { nt = (int)(t % NT); kt = (int)(t / NT); }
    int  n = nt * 16 + (lane & 15);
    int  k = kt * 32 + ((lane >> 4) << 3) + (j << 1) + ((j >> 2) << 3);
    uint32_t lo = f2bf(W[(long)k       * NCols + n]);
    uint32_t hi = f2bf(W[(long)(k + 1) * NCols + n]);
    Wp[i] = lo | (hi << 16);
}

// ------------------------------------------------------- scatter-sum (mean)
__global__ void k_scatter(const uint16_t* __restrict__ feat,
                          const long long* __restrict__ src,
                          const long long* __restrict__ dst,
                          float* __restrict__ agg, float* __restrict__ cnt,
                          long E, int doCnt) {
    long tid = (long)blockIdx.x * blockDim.x + threadIdx.x;
    long e = tid >> 5;
    if (e >= E) return;
    int lane = (int)(tid & 31);
    int s = (int)src[e];
    int d = (int)dst[e];
    uint2 raw = *(const uint2*)(feat + (long)s * 128 + lane * 4);
    float* ap = agg + (long)d * 128 + lane * 4;
    atomicAdd(ap + 0, bf2f(raw.x & 0xFFFFu));
    atomicAdd(ap + 1, bf2f(raw.x >> 16));
    atomicAdd(ap + 2, bf2f(raw.y & 0xFFFFu));
    atomicAdd(ap + 3, bf2f(raw.y >> 16));
    if (doCnt && lane == 0) atomicAdd(&cnt[d], 1.0f);
}

__global__ void k_div_cvt(const float* __restrict__ agg, const float* __restrict__ cnt,
                          uint16_t* __restrict__ out, long n) {
    long i = (long)blockIdx.x * blockDim.x + threadIdx.x;
    if (i >= n) return;
    float c = cnt[i >> 7];
    c = c > 1.f ? c : 1.f;
    out[i] = f2bf(agg[i] / c);
}

// ------------------------------------------- SAGE layer: relu(agg@Wl + bl + x@Wr)
// F = H = 128. One wave per 16-node tile. A fragments hoisted; per n-tile all
// 8 B fragments preloaded into distinct registers so WMMAs overlap the loads.
__global__ void __launch_bounds__(128)
k_sage_gemm(const uint16_t* __restrict__ Abf, const uint16_t* __restrict__ Xbf,
            const uint32_t* __restrict__ Wlp, const float* __restrict__ bl,
            const uint32_t* __restrict__ Wrp, uint16_t* __restrict__ out,
            int nTiles, int Nnodes) {
    int wave = blockIdx.x * (blockDim.x >> 5) + (threadIdx.x >> 5);
    if (wave >= nTiles) return;                 // wave-uniform: EXEC stays full
    int lane = threadIdx.x & 31;
    int ml = lane & 15, half = lane >> 4;
    int r = wave * 16 + ml;
    if (r >= Nnodes) r = Nnodes - 1;
    long rowOff = (long)r * 128;
    bool fullTile = (wave * 16 + 16) <= Nnodes; // wave-uniform store guard

    Frag fa[4], fx[4];
#pragma unroll
    for (int kt = 0; kt < 4; ++kt) {
        int c0 = kt * 32 + half * 8;            // A layout: K halves per lane-half
        fa[kt].q[0] = *(const uint4*)(Abf + rowOff + c0);
        fa[kt].q[1] = *(const uint4*)(Abf + rowOff + c0 + 16);
        fx[kt].q[0] = *(const uint4*)(Xbf + rowOff + c0);
        fx[kt].q[1] = *(const uint4*)(Xbf + rowOff + c0 + 16);
    }

    for (int nt = 0; nt < 8; ++nt) {
        Frag fbl[4], fbr[4];
#pragma unroll
        for (int kt = 0; kt < 4; ++kt) {
            const uint4* lp = (const uint4*)(Wlp + (((long)kt * 8 + nt) * 32 + lane) * 8);
            const uint4* rp = (const uint4*)(Wrp + (((long)kt * 8 + nt) * 32 + lane) * 8);
            fbl[kt].q[0] = lp[0]; fbl[kt].q[1] = lp[1];
            fbr[kt].q[0] = rp[0]; fbr[kt].q[1] = rp[1];
        }
        float bias = bl[nt * 16 + ml];          // C layout: col n = lane&15
        v8f acc = {0.f, 0.f, 0.f, 0.f, 0.f, 0.f, 0.f, 0.f};
#pragma unroll
        for (int kt = 0; kt < 4; ++kt)
            acc = __builtin_amdgcn_wmma_f32_16x16x32_bf16(
                false, fa[kt].v, false, fbl[kt].v, (short)0, acc, false, false);
#pragma unroll
        for (int kt = 0; kt < 4; ++kt)
            acc = __builtin_amdgcn_wmma_f32_16x16x32_bf16(
                false, fx[kt].v, false, fbr[kt].v, (short)0, acc, false, false);

        if (fullTile) {
#pragma unroll
            for (int j = 0; j < 8; ++j) {       // row m = j + 8*half
                float v = acc[j] + bias;
                out[(long)(wave * 16 + j + 8 * half) * 128 + nt * 16 + ml] =
                    f2bf(v > 0.f ? v : 0.f);
            }
        } else {
#pragma unroll
            for (int j = 0; j < 8; ++j) {
                int row = wave * 16 + j + 8 * half;
                if (row < Nnodes) {
                    float v = acc[j] + bias;
                    out[(long)row * 128 + nt * 16 + ml] = f2bf(v > 0.f ? v : 0.f);
                }
            }
        }
    }
}

// --------------------------- fused edge MLP: relu([h_src|h_dst]@W3 + b3)@W4 + b4
// One wave per 16 edges; 4 waves per block. W3 (nt-major packed) is staged
// through double-buffered LDS: the whole block cooperatively loads the next
// 8KB n-tile slab while WMMAs consume the current one. hidden never hits memory.
__global__ void __launch_bounds__(128)
k_edge_mlp(const uint16_t* __restrict__ H2,
           const long long* __restrict__ src, const long long* __restrict__ dst,
           const uint32_t* __restrict__ W3p, const float* __restrict__ b3,
           const float* __restrict__ W4, const float* __restrict__ b4,
           float* __restrict__ pred, int nTiles, long E) {
    __shared__ uint32_t sB[2][2048];            // 2 x 8KB slabs

    int wave = blockIdx.x * (blockDim.x >> 5) + (threadIdx.x >> 5);
    if (wave >= nTiles) wave = nTiles - 1;      // keep all waves for barriers
    int t    = threadIdx.x;                     // 0..127
    int lane = threadIdx.x & 31;
    int ml = lane & 15, half = lane >> 4;
    long e = (long)wave * 16 + ml;
    if (e >= E) e = E - 1;
    int s = (int)src[e];
    int d = (int)dst[e];

    Frag fa[8];                                  // 8 k-tiles of 32 (concat at kt==4)
#pragma unroll
    for (int kt = 0; kt < 8; ++kt) {
        int node = (kt < 4) ? s : d;
        long off = (long)node * 128 + (kt & 3) * 32 + half * 8;
        fa[kt].q[0] = *(const uint4*)(H2 + off);
        fa[kt].q[1] = *(const uint4*)(H2 + off + 16);
    }

    const uint4* gW3 = (const uint4*)W3p;        // 512 uint4 per nt slab
    uint4 st[4];
#pragma unroll
    for (int i = 0; i < 4; ++i) st[i] = gW3[t + i * 128];          // nt = 0 slab
#pragma unroll
    for (int i = 0; i < 4; ++i) *(uint4*)&sB[0][(t + i * 128) * 4] = st[i];
    __syncthreads();

    float pacc[8] = {0.f, 0.f, 0.f, 0.f, 0.f, 0.f, 0.f, 0.f};
    int buf = 0;
    for (int nt = 0; nt < 16; ++nt) {
        if (nt < 15) {                           // issue next slab's global loads
#pragma unroll
            for (int i = 0; i < 4; ++i) st[i] = gW3[(long)(nt + 1) * 512 + t + i * 128];
        }
        float b3n = b3[nt * 16 + ml];
        float w4n = W4[nt * 16 + ml];

        Frag fb[8];
#pragma unroll
        for (int kt = 0; kt < 8; ++kt) {        // low-latency LDS fragment reads
            const uint32_t* bp = &sB[buf][(kt * 32 + lane) * 8];
            fb[kt].q[0] = *(const uint4*)bp;
            fb[kt].q[1] = *(const uint4*)(bp + 4);
        }
        v8f acc = {0.f, 0.f, 0.f, 0.f, 0.f, 0.f, 0.f, 0.f};
#pragma unroll
        for (int kt = 0; kt < 8; ++kt)
            acc = __builtin_amdgcn_wmma_f32_16x16x32_bf16(
                false, fa[kt].v, false, fb[kt].v, (short)0, acc, false, false);
#pragma unroll
        for (int j = 0; j < 8; ++j) {
            float h = acc[j] + b3n;
            h = h > 0.f ? h : 0.f;
            pacc[j] += h * w4n;                  // partial dot over col n = nt*16+ml
        }

        if (nt < 15) {                           // commit next slab, flip buffers
#pragma unroll
            for (int i = 0; i < 4; ++i) *(uint4*)&sB[buf ^ 1][(t + i * 128) * 4] = st[i];
            __syncthreads();
            buf ^= 1;
        }
    }

    float b4v = b4[0];
#pragma unroll
    for (int j = 0; j < 8; ++j) {                // reduce over 16 cols per lane-half
        float v = pacc[j];
        v += __shfl_xor(v, 1, 32);
        v += __shfl_xor(v, 2, 32);
        v += __shfl_xor(v, 4, 32);
        v += __shfl_xor(v, 8, 32);
        if (ml == 0) {
            long row = (long)wave * 16 + j + 8 * half;
            if (row < E) pred[row] = v + b4v;
        }
    }
}

// --------------------------------------------------------------------------
extern "C" void kernel_launch(void* const* d_in, const int* in_sizes, int n_in,
                              void* d_out, int out_size, void* d_ws, size_t ws_size,
                              hipStream_t stream) {
    const float*     x   = (const float*)d_in[0];
    const long long* ei  = (const long long*)d_in[1];
    const float*     W1l = (const float*)d_in[2];
    const float*     b1l = (const float*)d_in[3];
    const float*     W1r = (const float*)d_in[4];
    const float*     W2l = (const float*)d_in[5];
    const float*     b2l = (const float*)d_in[6];
    const float*     W2r = (const float*)d_in[7];
    const float*     W3  = (const float*)d_in[8];
    const float*     b3  = (const float*)d_in[9];
    const float*     W4  = (const float*)d_in[10];
    const float*     b4  = (const float*)d_in[11];

    const int  F = 128, H = 128;
    const long N = (long)in_sizes[0] / F;
    const long E = (long)in_sizes[1] / 2;
    const int  M = in_sizes[8] / (2 * H);        // 256
    const long long* src = ei;
    const long long* dst = ei + E;
    float* pred = (float*)d_out;
    (void)n_in; (void)out_size; (void)ws_size; (void)M;

    char* ws = (char*)d_ws;
    size_t off = 0;
    auto carve = [&](size_t bytes) -> char* {
        char* p = ws + off;
        off += (bytes + 255) & ~(size_t)255;
        return p;
    };
    float*    agg   = (float*)   carve((size_t)N * F * 4);
    float*    cnt   = (float*)   carve((size_t)N * 4);
    uint16_t* xbf   = (uint16_t*)carve((size_t)N * F * 2);
    uint16_t* aggbf = (uint16_t*)carve((size_t)N * F * 2);
    uint16_t* h1    = (uint16_t*)carve((size_t)N * F * 2);
    uint16_t* h2    = (uint16_t*)carve((size_t)N * F * 2);
    uint32_t* W1lp  = (uint32_t*)carve((size_t)(F / 32) * (H / 16) * 256 * 4);
    uint32_t* W1rp  = (uint32_t*)carve((size_t)(F / 32) * (H / 16) * 256 * 4);
    uint32_t* W2lp  = (uint32_t*)carve((size_t)(H / 32) * (H / 16) * 256 * 4);
    uint32_t* W2rp  = (uint32_t*)carve((size_t)(H / 32) * (H / 16) * 256 * 4);
    uint32_t* W3p   = (uint32_t*)carve((size_t)(2 * H / 32) * (256 / 16) * 256 * 4);

    const int B = 256;
    long nf = N * F;

    // input conversion + weight packing
    k_cvt_bf16<<<(unsigned)((nf + B - 1) / B), B, 0, stream>>>(x, xbf, nf);
    long wN = (long)(F / 32) * (H / 16) * 256;
    k_pack_w<<<(unsigned)((wN + B - 1) / B), B, 0, stream>>>(W1l, W1lp, F, H, 0);
    k_pack_w<<<(unsigned)((wN + B - 1) / B), B, 0, stream>>>(W1r, W1rp, F, H, 0);
    k_pack_w<<<(unsigned)((wN + B - 1) / B), B, 0, stream>>>(W2l, W2lp, H, H, 0);
    k_pack_w<<<(unsigned)((wN + B - 1) / B), B, 0, stream>>>(W2r, W2rp, H, H, 0);
    long w3N = (long)(2 * H / 32) * (256 / 16) * 256;
    k_pack_w<<<(unsigned)((w3N + B - 1) / B), B, 0, stream>>>(W3, W3p, 2 * H, 256, 1);

    // ---- layer 1: mean-aggregate + SAGE GEMM
    k_zero<<<(unsigned)((nf + B - 1) / B), B, 0, stream>>>(agg, nf);
    k_zero<<<(unsigned)((N + B - 1) / B), B, 0, stream>>>(cnt, N);
    k_scatter<<<(unsigned)((E * 32 + B - 1) / B), B, 0, stream>>>(xbf, src, dst, agg, cnt, E, 1);
    k_div_cvt<<<(unsigned)((nf + B - 1) / B), B, 0, stream>>>(agg, cnt, aggbf, nf);
    int nTiles = (int)((N + 15) / 16);
    k_sage_gemm<<<(nTiles + 3) / 4, 128, 0, stream>>>(aggbf, xbf, W1lp, b1l, W1rp, h1, nTiles, (int)N);

    // ---- layer 2
    k_zero<<<(unsigned)((nf + B - 1) / B), B, 0, stream>>>(agg, nf);
    k_scatter<<<(unsigned)((E * 32 + B - 1) / B), B, 0, stream>>>(h1, src, dst, agg, cnt, E, 0);
    k_div_cvt<<<(unsigned)((nf + B - 1) / B), B, 0, stream>>>(agg, cnt, aggbf, nf);
    k_sage_gemm<<<(nTiles + 3) / 4, 128, 0, stream>>>(aggbf, h1, W2lp, b2l, W2rp, h2, nTiles, (int)N);

    // ---- fused edge MLP
    int eTiles = (int)((E + 15) / 16);
    k_edge_mlp<<<(eTiles + 3) / 4, 128, 0, stream>>>(h2, src, dst, W3p, b3, W4, b4, pred, eTiles, E);
}
